// MultiHeadedAttentionFast_51084341018811
// MI455X (gfx1250) — matrix operand order repaired and checked by reading, not compiled, not run on hardware
//
#include <hip/hip_runtime.h>
#include <cstdint>

// MultiHeadedAttention forward for B=4, S=2048, D=768, H=12, DH=64 on gfx1250.
// All matmuls via v_wmma_f32_16x16x32_f16 (wave32 WMMA), f32 accumulation.
// f32 operands are converted to f16 once in a pre-pass; GEMM inner loops are
// pure global_load_b128 + v_wmma. Attention stages K/mask tiles into LDS with
// GLOBAL_LOAD_ASYNC_TO_LDS_B128 (ASYNCcnt) when the builtin is available.

typedef __attribute__((ext_vector_type(16))) _Float16 v16h;
typedef __attribute__((ext_vector_type(8)))  _Float16 v8h;
typedef __attribute__((ext_vector_type(8)))  float    v8f;
typedef __attribute__((ext_vector_type(4)))  int      v4i;

static constexpr int Bn = 4;
static constexpr int Sn = 2048;
static constexpr int Dn = 768;
static constexpr int Hn = 12;
static constexpr int DH = 64;
static constexpr int Mrows = Bn * Sn;          // 8192

union v16h_u { v16h v; v8h h[2]; };

__device__ __forceinline__ v8f wmma16(v16h a, v16h b, v8f c) {
  return __builtin_amdgcn_wmma_f32_16x16x32_f16(false, a, false, b, (short)0, c,
                                                false, false);
}

// ---- async global->LDS staging (gfx1250 GLOBAL_LOAD_ASYNC_TO_LDS_B128) ------
#if defined(__has_builtin)
#if __has_builtin(__builtin_amdgcn_global_load_async_to_lds_b128)
#define HAVE_ASYNC_LDS 1
#endif
#endif

#ifdef HAVE_ASYNC_LDS
typedef __attribute__((address_space(1))) v4i gas_v4i;
typedef __attribute__((address_space(3))) v4i las_v4i;
__device__ __forceinline__ void async_ld16(const void* g, void* l) {
  __builtin_amdgcn_global_load_async_to_lds_b128(
      (gas_v4i*)(uintptr_t)g, (las_v4i*)(uint32_t)(uintptr_t)l, 0, 0);
}
__device__ __forceinline__ void async_wait0() {
#if __has_builtin(__builtin_amdgcn_s_wait_asynccnt)
  __builtin_amdgcn_s_wait_asynccnt(0);
#else
  asm volatile("s_wait_asynccnt 0x0" ::: "memory");
#endif
}
#endif

// ---- fragment loaders -------------------------------------------------------
// A-matrix 16x32 f16: lane m=lane&15; element j of v16h is
// K = j + (lane>=16?16:0)  -> one contiguous 32B group.
__device__ __forceinline__ v16h load_a_f16(const _Float16* base, int ldk, int kc) {
  int lane = threadIdx.x & 31;
  int off = (lane >> 4) ? 16 : 0;
  const _Float16* p = base + (size_t)(lane & 15) * ldk + kc + off;
  v16h_u u;
  u.h[0] = *(const v8h*)(p);
  u.h[1] = *(const v8h*)(p + 8);
  return u.v;
}

// B-matrix 32x16 f16: lane holds column n=lane&15; K = (lane>=16?16:0)+j contiguous.
__device__ __forceinline__ v16h load_b_f16(const _Float16* col_base, int ldk, int kc) {
  int lane = threadIdx.x & 31;
  const _Float16* p = col_base + (size_t)(lane & 15) * ldk + kc + ((lane >> 4) ? 16 : 0);
  v16h_u u;
  u.h[0] = *(const v8h*)(p);
  u.h[1] = *(const v8h*)(p + 8);
  return u.v;
}

__device__ __forceinline__ v16h load_b_lds(const _Float16* p16) {
  v16h_u u;
  u.h[0] = *(const v8h*)(p16);
  u.h[1] = *(const v8h*)(p16 + 8);
  return u.v;
}

// ---- kernel 0: bulk f32 -> f16 conversion (8 elements / thread) -------------
__global__ __launch_bounds__(256) void cvt_f16_kernel(
    const float* __restrict__ src, _Float16* __restrict__ dst, int n8) {
  int i = blockIdx.x * 256 + threadIdx.x;
  if (i >= n8) return;
  const float4* s = (const float4*)src + (size_t)2 * i;
  float4 a = s[0], b = s[1];
  v8h o;
  o[0] = (_Float16)a.x; o[1] = (_Float16)a.y;
  o[2] = (_Float16)a.z; o[3] = (_Float16)a.w;
  o[4] = (_Float16)b.x; o[5] = (_Float16)b.y;
  o[6] = (_Float16)b.z; o[7] = (_Float16)b.w;
  *(v8h*)(dst + (size_t)8 * i) = o;
}

// ---- kernel 1: fused QKV projection  out_f16 = f16((X @ W^T + b) * scale) ---
// 32x64 per-wave tile; grid.z = {Q,K,V}; Q is pre-scaled by 1/sqrt(DH)=0.125.
__global__ __launch_bounds__(256) void qkv_gemm_kernel(
    const _Float16* __restrict__ xh, const _Float16* __restrict__ keysh,
    const _Float16* __restrict__ valsh,
    const _Float16* __restrict__ Wqh, const float* __restrict__ bq,
    const _Float16* __restrict__ Wkh, const float* __restrict__ bk,
    const _Float16* __restrict__ Wvh, const float* __restrict__ bv,
    _Float16* __restrict__ qh, _Float16* __restrict__ kh,
    _Float16* __restrict__ vh) {
  const _Float16* A;  const _Float16* W;  const float* bias;  _Float16* out;
  float oscale;
  if (blockIdx.z == 0)      { A = xh;    W = Wqh; bias = bq; out = qh; oscale = 0.125f; }
  else if (blockIdx.z == 1) { A = keysh; W = Wkh; bias = bk; out = kh; oscale = 1.0f; }
  else                      { A = valsh; W = Wvh; bias = bv; out = vh; oscale = 1.0f; }

  int wid = blockIdx.x * 8 + (threadIdx.x >> 5);   // 0..3071
  int mtile = wid / (Dn / 64);                     // 0..255
  int ngrp  = wid % (Dn / 64);
  int mbase = mtile * 32, nbase = ngrp * 64;

  v8f acc[2][4] = {};
  const _Float16* A0 = A + (size_t)mbase * Dn;
  const _Float16* A1 = A + (size_t)(mbase + 16) * Dn;
  for (int kc = 0; kc < Dn; kc += 32) {
    v16h a0 = load_a_f16(A0, Dn, kc);
    v16h a1 = load_a_f16(A1, Dn, kc);
#pragma unroll
    for (int t = 0; t < 4; ++t) {
      v16h bfr = load_b_f16(W + (size_t)(nbase + 16 * t) * Dn, Dn, kc);
      acc[0][t] = wmma16(a0, bfr, acc[0][t]);
      acc[1][t] = wmma16(a1, bfr, acc[1][t]);
    }
  }
  int lane = threadIdx.x & 31, half = lane >> 4, nl = lane & 15;
#pragma unroll
  for (int t = 0; t < 4; ++t) {
    int col = nbase + 16 * t + nl;
    float bv0 = bias[col];
#pragma unroll
    for (int mi = 0; mi < 2; ++mi) {
#pragma unroll
      for (int r = 0; r < 8; ++r) {
        int row = mbase + 16 * mi + r + 8 * half;
        out[(size_t)row * Dn + col] = (_Float16)((acc[mi][t][r] + bv0) * oscale);
      }
    }
  }
}

// ---- kernel 2: flash attention ---------------------------------------------
// B*H*(S/128) blocks of 8 waves; each wave owns 16 queries x DH=64.
__global__ __launch_bounds__(256) void attn_kernel(
    const _Float16* __restrict__ qh, const _Float16* __restrict__ kh,
    const _Float16* __restrict__ vh, const unsigned char* __restrict__ mask,
    _Float16* __restrict__ ah) {
  __shared__ _Float16 kbuf[32 * 64];          // 32 keys x 64 dh, row-major
  __shared__ _Float16 vbufT[64 * 32];         // transposed: [dh][key]
  __shared__ unsigned char mbuf[128 * 32];    // mask tile [q_local][k_local]
  __shared__ _Float16 pbuf[8 * 16 * 32];      // per-wave P bounce (16x32)

  int bid = blockIdx.x;
  int qblk = bid & 15;                        // S/128 = 16
  int bh = bid >> 4;
  int b = bh / Hn, h = bh % Hn;
  int tid = threadIdx.x, w = tid >> 5, lane = tid & 31;
  int half = lane >> 4, nl = lane & 15;
  int q0 = qblk * 128, qbase = q0 + w * 16;
  int aoff = half ? 16 : 0;

  // Q A-fragments (already scaled by 1/sqrt(DH) at projection time)
  const _Float16* qtile = qh + ((size_t)(b * Sn + qbase) * Dn + h * DH);
  v16h aq[2];
#pragma unroll
  for (int c = 0; c < 2; ++c) aq[c] = load_a_f16(qtile, Dn, 32 * c);

  v8f acc[4] = {};
  float mrow[8], lrow[8];
#pragma unroll
  for (int r = 0; r < 8; ++r) { mrow[r] = -1e30f; lrow[r] = 0.0f; }

  for (int kc = 0; kc < Sn; kc += 32) {
    // ---- cooperative staging ----
    int row = tid >> 3, c8 = tid & 7;         // 32 rows x 8 chunks of 16B
    const _Float16* ks = kh + ((size_t)(b * Sn + kc + row) * Dn + h * DH + c8 * 8);
    int qr = tid >> 1, hh = tid & 1;          // 128 q-rows x 2 chunks of 16B
    const unsigned char* ms = mask + (size_t)(q0 + qr) * Sn + kc + hh * 16;
#ifdef HAVE_ASYNC_LDS
    async_ld16(ks, &kbuf[row * 64 + c8 * 8]);
    async_ld16(ms, &mbuf[qr * 32 + hh * 16]);
#else
    *(v8h*)&kbuf[row * 64 + c8 * 8] = *(const v8h*)ks;
    *(uint4*)&mbuf[qr * 32 + hh * 16] = *(const uint4*)ms;
#endif
    // V tile transpose goes through registers
    const _Float16* vs = vh + ((size_t)(b * Sn + kc + row) * Dn + h * DH + c8 * 8);
    v8h vv = *(const v8h*)vs;
#pragma unroll
    for (int e = 0; e < 8; ++e) vbufT[(c8 * 8 + e) * 32 + row] = vv[e];
#ifdef HAVE_ASYNC_LDS
    async_wait0();
#endif
    __syncthreads();

    // ---- scores: 16 queries x 32 keys via 4 WMMAs ----
    v8f s0 = {}, s1 = {};
#pragma unroll
    for (int c = 0; c < 2; ++c) {
      const _Float16* p0 = &kbuf[(0  + nl) * 64 + 32 * c + (half ? 16 : 0)];
      const _Float16* p1 = &kbuf[(16 + nl) * 64 + 32 * c + (half ? 16 : 0)];
      s0 = wmma16(aq[c], load_b_lds(p0), s0);
      s1 = wmma16(aq[c], load_b_lds(p1), s1);
    }

    // ---- masked online softmax (C-layout row M = r + 8*half) ----
    int w16 = w * 16;
#pragma unroll
    for (int r = 0; r < 8; ++r) {
      int ql = r + 8 * half;
      float x0 = s0[r], x1 = s1[r];
      if (!mbuf[(w16 + ql) * 32 + nl])      x0 = -10000.0f;
      if (!mbuf[(w16 + ql) * 32 + 16 + nl]) x1 = -10000.0f;
      float rm = fmaxf(x0, x1);
#pragma unroll
      for (int d = 8; d >= 1; d >>= 1) rm = fmaxf(rm, __shfl_xor(rm, d, 32));
      float mnew = fmaxf(mrow[r], rm);
      float scl = __expf(mrow[r] - mnew);
      float p0 = __expf(x0 - mnew);
      float p1 = __expf(x1 - mnew);
      float rs = p0 + p1;
#pragma unroll
      for (int d = 8; d >= 1; d >>= 1) rs += __shfl_xor(rs, d, 32);
      lrow[r] = lrow[r] * scl + rs;
      mrow[r] = mnew;
#pragma unroll
      for (int t = 0; t < 4; ++t) acc[t][r] *= scl;
      pbuf[w * 512 + ql * 32 + nl]      = (_Float16)p0;
      pbuf[w * 512 + ql * 32 + 16 + nl] = (_Float16)p1;
    }
    asm volatile("s_wait_dscnt 0" ::: "memory");  // same-wave DS fence

    // P as A-fragment (16x32)
    v16h pa;
    {
      const _Float16* pp = &pbuf[w * 512 + nl * 32 + aoff];
      v16h_u u;
      u.h[0] = *(const v8h*)(pp);
      u.h[1] = *(const v8h*)(pp + 8);
      pa = u.v;
    }
    // acc += P @ V
#pragma unroll
    for (int t = 0; t < 4; ++t) {
      const _Float16* vp = &vbufT[(16 * t + nl) * 32 + (half ? 16 : 0)];
      acc[t] = wmma16(pa, load_b_lds(vp), acc[t]);
    }
    __syncthreads();
  }

#pragma unroll
  for (int r = 0; r < 8; ++r) {
    float inv = 1.0f / lrow[r];
    int q = qbase + r + 8 * half;
#pragma unroll
    for (int t = 0; t < 4; ++t) {
      ah[(size_t)(b * Sn + q) * Dn + h * DH + 16 * t + nl] =
          (_Float16)(acc[t][r] * inv);
    }
  }
}

// ---- kernel 3: output projection  out_f32 = ah @ Wo^T + bo ------------------
__global__ __launch_bounds__(256) void out_proj_kernel(
    const _Float16* __restrict__ ah, const _Float16* __restrict__ Woh,
    const float* __restrict__ bo, float* __restrict__ out) {
  int wid = blockIdx.x * 8 + (threadIdx.x >> 5);
  int mtile = wid / (Dn / 64);
  int ngrp  = wid % (Dn / 64);
  int mbase = mtile * 32, nbase = ngrp * 64;

  v8f acc[2][4] = {};
  const _Float16* A0 = ah + (size_t)mbase * Dn;
  const _Float16* A1 = ah + (size_t)(mbase + 16) * Dn;
  for (int kc = 0; kc < Dn; kc += 32) {
    v16h a0 = load_a_f16(A0, Dn, kc);
    v16h a1 = load_a_f16(A1, Dn, kc);
#pragma unroll
    for (int t = 0; t < 4; ++t) {
      v16h bfr = load_b_f16(Woh + (size_t)(nbase + 16 * t) * Dn, Dn, kc);
      acc[0][t] = wmma16(a0, bfr, acc[0][t]);
      acc[1][t] = wmma16(a1, bfr, acc[1][t]);
    }
  }
  int lane = threadIdx.x & 31, half = lane >> 4, nl = lane & 15;
#pragma unroll
  for (int t = 0; t < 4; ++t) {
    int col = nbase + 16 * t + nl;
    float bv0 = bo[col];
#pragma unroll
    for (int mi = 0; mi < 2; ++mi) {
#pragma unroll
      for (int r = 0; r < 8; ++r) {
        int rowi = mbase + 16 * mi + r + 8 * half;
        out[(size_t)rowi * Dn + col] = acc[mi][t][r] + bv0;
      }
    }
  }
}

extern "C" void kernel_launch(void* const* d_in, const int* in_sizes, int n_in,
                              void* d_out, int out_size, void* d_ws,
                              size_t ws_size, hipStream_t stream) {
  const float* x      = (const float*)d_in[0];
  const float* keys   = (const float*)d_in[1];
  const float* values = (const float*)d_in[2];
  const unsigned char* mask = (const unsigned char*)d_in[3];
  const float* Wq = (const float*)d_in[4];
  const float* bq = (const float*)d_in[5];
  const float* Wk = (const float*)d_in[6];
  const float* bk = (const float*)d_in[7];
  const float* Wv = (const float*)d_in[8];
  const float* bv = (const float*)d_in[9];
  const float* Wo = (const float*)d_in[10];
  const float* bo = (const float*)d_in[11];
  float* out = (float*)d_out;

  size_t MD = (size_t)Mrows * Dn;        // 6,291,456
  size_t DD = (size_t)Dn * Dn;           // 589,824
  _Float16* p = (_Float16*)d_ws;
  _Float16* qh  = p;  p += MD;
  _Float16* khb = p;  p += MD;
  _Float16* vhb = p;  p += MD;
  _Float16* ahb = p;  p += MD;
  _Float16* xh  = p;  p += MD;
  _Float16* keysh = p; p += MD;
  _Float16* valsh = p; p += MD;
  _Float16* Wqh = p;  p += DD;
  _Float16* Wkh = p;  p += DD;
  _Float16* Wvh = p;  p += DD;
  _Float16* Woh = p;  p += DD;           // ~93 MB total f16 workspace

  // pre-pass: convert operands to f16 once
  int n8m = (int)(MD / 8), n8d = (int)(DD / 8);
  int gbm = (n8m + 255) / 256, gbd = (n8d + 255) / 256;
  cvt_f16_kernel<<<gbm, 256, 0, stream>>>(x, xh, n8m);
  cvt_f16_kernel<<<gbm, 256, 0, stream>>>(keys, keysh, n8m);
  cvt_f16_kernel<<<gbm, 256, 0, stream>>>(values, valsh, n8m);
  cvt_f16_kernel<<<gbd, 256, 0, stream>>>(Wq, Wqh, n8d);
  cvt_f16_kernel<<<gbd, 256, 0, stream>>>(Wk, Wkh, n8d);
  cvt_f16_kernel<<<gbd, 256, 0, stream>>>(Wv, Wvh, n8d);
  cvt_f16_kernel<<<gbd, 256, 0, stream>>>(Wo, Woh, n8d);

  int tiles = (Mrows / 32) * (Dn / 64);  // 3072 wave-tiles
  int blocks = tiles / 8;                // 384
  qkv_gemm_kernel<<<dim3(blocks, 1, 3), 256, 0, stream>>>(
      xh, keysh, valsh, Wqh, bq, Wkh, bk, Wvh, bv, qh, khb, vhb);
  attn_kernel<<<Bn * Hn * (Sn / 128), 256, 0, stream>>>(qh, khb, vhb, mask, ahb);
  out_proj_kernel<<<blocks, 256, 0, stream>>>(ahb, Woh, bo, out);
}
